// GPTLanguageModel_78769700208872
// MI455X (gfx1250) — matrix operand order repaired and checked by reading, not compiled, not run on hardware
//
#include <hip/hip_runtime.h>
#include <hip/hip_bf16.h>

typedef __attribute__((ext_vector_type(16))) _Float16 v16h;
typedef __attribute__((ext_vector_type(8)))  _Float16 v8h;
typedef __attribute__((ext_vector_type(8)))  float    v8f;

// ---- model dims ----
constexpr int Bb = 2, Tt = 1024, BT = 2048, Dd = 512, Hh = 8, HDk = 64;
constexpr int Ll = 6, Vv = 32000, FF = 2048;
constexpr float EPSf = 1e-5f;

// ---- gemm tiling: block tile 64x128, 8 waves of 32, each wave 32x32 (2x2 wmma 16x16x32) ----
constexpr int BM = 64, BN = 128, BK = 32, KPAD = 40;

#define DEVI __device__ __forceinline__

// ---- CDNA5 async memory->LDS copy (ASYNCcnt path, ISA 15.18 op 98) ----
DEVI void async_copy_b128(void* lds_dst, const void* gsrc) {
  asm volatile("global_load_async_to_lds_b128 %0, %1, off"
               :: "v"((unsigned)(uintptr_t)lds_dst), "v"(gsrc)
               : "memory");
}
DEVI void async_wait0() { asm volatile("s_wait_asynccnt 0" ::: "memory"); }
DEVI void ds_wait0()    { asm volatile("s_wait_dscnt 0" ::: "memory"); }

// A-fragment (16x32 f16): lane<16 -> row=lane, K 0..7 & 16..23 ; lane>=16 -> row=lane-16, K 8..15 & 24..31
DEVI v16h frag_a(const _Float16* s, int stride, int m0, int kbase, int lane) {
  const _Float16* p = s + (m0 + (lane & 15)) * stride + kbase + ((lane >> 4) << 3);
  v8h lo = *(const v8h*)p;
  v8h hi = *(const v8h*)(p + 16);
  v16h r;
#pragma unroll
  for (int i = 0; i < 8; ++i) { r[i] = lo[i]; r[i + 8] = hi[i]; }
  return r;
}

// B-fragment (32x16 f16) from LDS stored as [n][k]: lane<16 -> col n = lane, K 0..7 & 16..23 ; lane>=16 -> K 8..15 & 24..31
DEVI v16h frag_b(const _Float16* s, int stride, int n0, int kbase, int lane) {
  const _Float16* p = s + (n0 + (lane & 15)) * stride + kbase + ((lane >> 4) << 3);
  v8h lo = *(const v8h*)p;
  v8h hi = *(const v8h*)(p + 16);
  v16h r;
#pragma unroll
  for (int i = 0; i < 8; ++i) { r[i] = lo[i]; r[i + 8] = hi[i]; }
  return r;
}

#define WMMA_F16(a, b, c) \
  __builtin_amdgcn_wmma_f32_16x16x32_f16(false, (a), false, (b), (short)0, (c), false, false)

// ---------------- generic WMMA GEMM: out = act(A[f16,MxK] * W[f32,KxN] + bias) (+residual) ----------------
template <bool RELU>
__global__ __launch_bounds__(256) void gemm_kernel(
    const _Float16* __restrict__ A, int lda,
    const float* __restrict__ Wp, int ldw,
    const float* __restrict__ bias,
    const float* __restrict__ resid,
    float* __restrict__ C32,
    _Float16* __restrict__ C16, int ldc,
    int M, int N, int K)
{
  __shared__ __align__(16) _Float16 sA[BM * KPAD];
  __shared__ __align__(16) _Float16 sWT[BN * KPAD];

  const int t = threadIdx.x;
  const int lane = t & 31;
  const int wv = t >> 5;
  const int wy = wv >> 2, wx = wv & 3;
  const int m0 = blockIdx.y * BM;
  const int n0 = blockIdx.x * BN;

  v8f acc[2][2] = {};

  const int ar = t >> 2;        // A tile row 0..63
  const int ac = (t & 3) * 8;   // A tile col (8 halves)

  for (int k0 = 0; k0 < K; k0 += BK) {
    // ---- stage A tile (64x32 f16) via async LDS copy ----
    async_copy_b128(sA + ar * KPAD + ac,
                    A + (long)(m0 + ar) * lda + k0 + ac);
    if (k0 + BK < K)
      __builtin_prefetch((const void*)(A + (long)(m0 + ar) * lda + k0 + BK + ac), 0, 0);

    // ---- stage W tile transposed into sWT[n][k], f32 -> f16 on the fly ----
    {
      const int wrb = t >> 5;          // 0..7
      const int wc = (t & 31) * 4;     // 0..124
#pragma unroll
      for (int j = 0; j < 4; ++j) {
        const int kr = wrb + j * 8;    // 0..31
        const long rowoff = (long)(k0 + kr) * ldw + n0 + wc;
        float x0, x1, x2, x3;
        if (n0 + wc + 4 <= N) {
          float4 vq = *(const float4*)(Wp + rowoff);
          x0 = vq.x; x1 = vq.y; x2 = vq.z; x3 = vq.w;
        } else {
          x0 = (n0 + wc + 0 < N) ? Wp[rowoff + 0] : 0.f;
          x1 = (n0 + wc + 1 < N) ? Wp[rowoff + 1] : 0.f;
          x2 = (n0 + wc + 2 < N) ? Wp[rowoff + 2] : 0.f;
          x3 = (n0 + wc + 3 < N) ? Wp[rowoff + 3] : 0.f;
        }
        sWT[(wc + 0) * KPAD + kr] = (_Float16)x0;
        sWT[(wc + 1) * KPAD + kr] = (_Float16)x1;
        sWT[(wc + 2) * KPAD + kr] = (_Float16)x2;
        sWT[(wc + 3) * KPAD + kr] = (_Float16)x3;
      }
    }
    async_wait0();
    __syncthreads();

    const v16h a0 = frag_a(sA, KPAD, wy * 32 + 0, 0, lane);
    const v16h a1 = frag_a(sA, KPAD, wy * 32 + 16, 0, lane);
    const v16h b0 = frag_b(sWT, KPAD, wx * 32 + 0, 0, lane);
    const v16h b1 = frag_b(sWT, KPAD, wx * 32 + 16, 0, lane);
    acc[0][0] = WMMA_F16(a0, b0, acc[0][0]);
    acc[0][1] = WMMA_F16(a0, b1, acc[0][1]);
    acc[1][0] = WMMA_F16(a1, b0, acc[1][0]);
    acc[1][1] = WMMA_F16(a1, b1, acc[1][1]);
    __syncthreads();
  }

  // ---- epilogue: C/D layout -> VGPR r: M = r + (lane>=16 ? 8:0), N = lane%16 ----
  const int lh = (lane >> 4) * 8;
  const int ln = lane & 15;
#pragma unroll
  for (int mi = 0; mi < 2; ++mi)
#pragma unroll
    for (int ni = 0; ni < 2; ++ni) {
      const int nn = n0 + wx * 32 + ni * 16 + ln;
      if (nn >= N) continue;
      const float bval = bias ? bias[nn] : 0.f;
#pragma unroll
      for (int r = 0; r < 8; ++r) {
        const int mm = m0 + wy * 32 + mi * 16 + lh + r;
        float v = acc[mi][ni][r] + bval;
        if constexpr (RELU) v = fmaxf(v, 0.f);
        const long ci = (long)mm * ldc + nn;
        if (resid) v += resid[ci];
        if (C32) C32[ci] = v;
        if (C16) C16[ci] = (_Float16)v;
      }
    }
}

// ---------------- fused flash attention (causal, online softmax), WMMA for QK^T and P.V ----------------
// Block: 128 threads = 4 waves; each wave owns 16 query rows of a 64-row q-tile; loops keys 32 at a time.
constexpr int QS = 72;   // sQ/sK row stride (halves): 144B rows, 16B aligned, conflict-free
constexpr int VS = 40;   // sVT row stride
constexpr int PS = 40;   // sP row stride

__global__ __launch_bounds__(128) void flash_attn_kernel(
    const _Float16* __restrict__ q16, const _Float16* __restrict__ k16,
    const _Float16* __restrict__ v16, _Float16* __restrict__ attn16, float scale)
{
  __shared__ __align__(16) _Float16 sQ[64 * QS];
  __shared__ __align__(16) _Float16 sK[32 * QS];
  __shared__ __align__(16) _Float16 sVT[64 * VS];
  __shared__ __align__(16) _Float16 sP[4 * 16 * PS];

  const int t = threadIdx.x;
  const int lane = t & 31;
  const int w = t >> 5;                 // wave 0..3
  const int bh = blockIdx.y;
  const int b = bh >> 3, h = bh & 7;
  const int q0 = blockIdx.x * 64;

  const _Float16* qb = q16 + (long)b * Tt * Dd + h * HDk;
  const _Float16* kb = k16 + (long)b * Tt * Dd + h * HDk;
  const _Float16* vb = v16 + (long)b * Tt * Dd + h * HDk;

  // stage Q tile 64x64 (async): 512 chunks of 16B / 128 threads
  for (int c = t; c < 64 * 8; c += 128) {
    const int r = c >> 3, col = (c & 7) * 8;
    async_copy_b128(sQ + r * QS + col, qb + (long)(q0 + r) * Dd + col);
  }

  v8f oacc[4] = {};
  float mrow[8], lrow[8];
#pragma unroll
  for (int r = 0; r < 8; ++r) { mrow[r] = -__builtin_inff(); lrow[r] = 0.f; }

  const int lh = (lane >> 4) * 8;
  const int ln = lane & 15;
  const int qhi = q0 + w * 16 + 15;     // highest query row this wave owns

  for (int kt = 0; kt < q0 + 64; kt += 32) {
    // stage K tile 32x64 (async) -- natural [key][d] layout == B-fragment layout for QK^T
    for (int c = t; c < 32 * 8; c += 128) {
      const int r = c >> 3, col = (c & 7) * 8;
      async_copy_b128(sK + r * QS + col, kb + (long)(kt + r) * Dd + col);
    }
    // stage V tile transposed -> sVT[d][key]
    {
      const int r = t >> 2, col = (t & 3) * 16;
      const _Float16* src = vb + (long)(kt + r) * Dd + col;
      v8h x0 = *(const v8h*)src;
      v8h x1 = *(const v8h*)(src + 8);
#pragma unroll
      for (int j = 0; j < 8; ++j) {
        sVT[(col + j) * VS + r]     = x0[j];
        sVT[(col + 8 + j) * VS + r] = x1[j];
      }
    }
    async_wait0();
    __syncthreads();

    if (kt <= qhi) {  // wave-uniform: EXEC stays all-ones for WMMA
      // S(16q x 32k) = Q . K^T over HD=64 (2 chunks)
      v8f sacc[2] = {};
#pragma unroll
      for (int kc = 0; kc < HDk; kc += 32) {
        const v16h aq  = frag_a(sQ, QS, w * 16, kc, lane);
        const v16h bk0 = frag_b(sK, QS, 0,  kc, lane);
        const v16h bk1 = frag_b(sK, QS, 16, kc, lane);
        sacc[0] = WMMA_F16(aq, bk0, sacc[0]);
        sacc[1] = WMMA_F16(aq, bk1, sacc[1]);
      }
      // online softmax: row reductions across 16 lanes (xor 1/2/4/8 stays within lane halves)
      float p0v[8], p1v[8];
#pragma unroll
      for (int r = 0; r < 8; ++r) {
        const int qq = q0 + w * 16 + lh + r;
        float s0 = sacc[0][r] * scale; if (kt + ln > qq)      s0 = -__builtin_inff();
        float s1 = sacc[1][r] * scale; if (kt + 16 + ln > qq) s1 = -__builtin_inff();
        float mx = fmaxf(s0, s1);
        mx = fmaxf(mx, __shfl_xor(mx, 1));
        mx = fmaxf(mx, __shfl_xor(mx, 2));
        mx = fmaxf(mx, __shfl_xor(mx, 4));
        mx = fmaxf(mx, __shfl_xor(mx, 8));
        const float mnew  = fmaxf(mrow[r], mx);
        const float alpha = __expf(mrow[r] - mnew);
        const float p0 = __expf(s0 - mnew);
        const float p1 = __expf(s1 - mnew);
        float sum = p0 + p1;
        sum += __shfl_xor(sum, 1);
        sum += __shfl_xor(sum, 2);
        sum += __shfl_xor(sum, 4);
        sum += __shfl_xor(sum, 8);
        lrow[r] = lrow[r] * alpha + sum;
        mrow[r] = mnew;
        p0v[r] = p0; p1v[r] = p1;
#pragma unroll
        for (int nj = 0; nj < 4; ++nj) oacc[nj][r] *= alpha;
      }
      // P (C-layout) -> per-wave LDS -> A-fragment layout
      _Float16* sPw = sP + w * 16 * PS;
#pragma unroll
      for (int r = 0; r < 8; ++r) {
        sPw[(lh + r) * PS + ln]      = (_Float16)p0v[r];
        sPw[(lh + r) * PS + 16 + ln] = (_Float16)p1v[r];
      }
      ds_wait0();
      const v16h ap = frag_a(sPw, PS, 0, 0, lane);
#pragma unroll
      for (int nj = 0; nj < 4; ++nj) {
        const v16h bv = frag_b(sVT, VS, nj * 16, 0, lane);
        oacc[nj] = WMMA_F16(ap, bv, oacc[nj]);
      }
    }
    __syncthreads();
  }

  // normalize and write attn (f16, [B,T,D] with head offset)
  _Float16* ob = attn16 + (long)b * Tt * Dd + h * HDk;
#pragma unroll
  for (int r = 0; r < 8; ++r) {
    const float inv = 1.f / lrow[r];
    const int qq = q0 + w * 16 + lh + r;
#pragma unroll
    for (int nj = 0; nj < 4; ++nj)
      ob[(long)qq * Dd + nj * 16 + ln] = (_Float16)(oacc[nj][r] * inv);
  }
}

// ---------------- layernorm over D=512, f16 out ----------------
__global__ __launch_bounds__(256) void layernorm_kernel(
    const float* __restrict__ x, const float* __restrict__ g,
    const float* __restrict__ bt, _Float16* __restrict__ out)
{
  __shared__ float red[256];
  const long row = blockIdx.x;
  const int t = threadIdx.x;
  const float* xr = x + row * Dd;
  const float a0 = xr[t], a1 = xr[t + 256];
  red[t] = a0 + a1; __syncthreads();
  for (int o = 128; o > 0; o >>= 1) { if (t < o) red[t] += red[t + o]; __syncthreads(); }
  const float mu = red[0] * (1.f / Dd); __syncthreads();
  const float d0 = a0 - mu, d1 = a1 - mu;
  red[t] = d0 * d0 + d1 * d1; __syncthreads();
  for (int o = 128; o > 0; o >>= 1) { if (t < o) red[t] += red[t + o]; __syncthreads(); }
  const float rs = rsqrtf(red[0] * (1.f / Dd) + EPSf);
  out[row * Dd + t]       = (_Float16)(d0 * rs * g[t] + bt[t]);
  out[row * Dd + t + 256] = (_Float16)(d1 * rs * g[t + 256] + bt[t + 256]);
}

// ---------------- embedding gather ----------------
__global__ __launch_bounds__(256) void embed_kernel(
    const int* __restrict__ idx, const float* __restrict__ tok,
    const float* __restrict__ pos, float* __restrict__ x)
{
  const long row = blockIdx.x;
  const int t = threadIdx.x;
  const long tokid = idx[row];
  const long tpos = row % Tt;
  x[row * Dd + t]       = tok[tokid * Dd + t] + pos[tpos * Dd + t];
  x[row * Dd + t + 256] = tok[tokid * Dd + t + 256] + pos[tpos * Dd + t + 256];
}

// ---------------- loss = -mean(log_softmax(logits)[tgt]) ----------------
__global__ void zero_loss_kernel(float* p) { *p = 0.f; }

__global__ __launch_bounds__(256) void loss_kernel(
    const float* __restrict__ logits, const int* __restrict__ tgt, float* __restrict__ loss)
{
  __shared__ float red[256];
  const long row = blockIdx.x;
  const float* l = logits + row * (long)Vv;
  const int t = threadIdx.x;
  float mx = -__builtin_inff();
  for (int i = t; i < Vv; i += 256) mx = fmaxf(mx, l[i]);
  red[t] = mx; __syncthreads();
  for (int o = 128; o > 0; o >>= 1) { if (t < o) red[t] = fmaxf(red[t], red[t + o]); __syncthreads(); }
  mx = red[0]; __syncthreads();
  float s = 0.f;
  for (int i = t; i < Vv; i += 256) s += expf(l[i] - mx);
  red[t] = s; __syncthreads();
  for (int o = 128; o > 0; o >>= 1) { if (t < o) red[t] += red[t + o]; __syncthreads(); }
  if (t == 0) {
    const float lp = l[tgt[row]] - mx - logf(red[0]);
    atomicAdd(loss, -lp / (float)BT);
  }
}

// ---------------- driver ----------------
extern "C" void kernel_launch(void* const* d_in, const int* in_sizes, int n_in,
                              void* d_out, int out_size, void* d_ws, size_t ws_size,
                              hipStream_t stream)
{
  (void)in_sizes; (void)n_in; (void)out_size; (void)ws_size;
  const int*   idx    = (const int*)d_in[0];
  const int*   tgts   = (const int*)d_in[1];
  const float* tok    = (const float*)d_in[2];
  const float* pos    = (const float*)d_in[3];
  const float* Wq     = (const float*)d_in[4];
  const float* Wk     = (const float*)d_in[5];
  const float* Wvv    = (const float*)d_in[6];
  const float* Wo     = (const float*)d_in[7];
  const float* bo     = (const float*)d_in[8];
  const float* W1     = (const float*)d_in[9];
  const float* b1     = (const float*)d_in[10];
  const float* W2     = (const float*)d_in[11];
  const float* b2     = (const float*)d_in[12];
  const float* ln1g   = (const float*)d_in[13];
  const float* ln1b   = (const float*)d_in[14];
  const float* ln2g   = (const float*)d_in[15];
  const float* ln2b   = (const float*)d_in[16];
  const float* lnfg   = (const float*)d_in[17];
  const float* lnfb   = (const float*)d_in[18];
  const float* Whead  = (const float*)d_in[19];
  const float* bhead  = (const float*)d_in[20];

  char* ws = (char*)d_ws;
  size_t off = 0;
  auto carve = [&](size_t bytes) -> void* {
    void* p = ws + off;
    off += (bytes + 255) & ~(size_t)255;
    return p;
  };
  float*    x      = (float*)carve((size_t)BT * Dd * 4);
  _Float16* h16    = (_Float16*)carve((size_t)BT * Dd * 2);
  _Float16* q16    = (_Float16*)carve((size_t)BT * Dd * 2);
  _Float16* k16    = (_Float16*)carve((size_t)BT * Dd * 2);
  _Float16* v16    = (_Float16*)carve((size_t)BT * Dd * 2);
  _Float16* attn16 = (_Float16*)carve((size_t)BT * Dd * 2);
  _Float16* mlp16  = (_Float16*)carve((size_t)BT * FF * 2);

  float* logits = (float*)d_out;
  float* loss = logits + (long)BT * Vv;

  const dim3 blk(256);
  const float scale = 0.044194173824159216f;  // 1/sqrt(512) (scale by n_embd, per reference)

  embed_kernel<<<BT, blk, 0, stream>>>(idx, tok, pos, x);

  for (int l = 0; l < Ll; ++l) {
    layernorm_kernel<<<BT, blk, 0, stream>>>(x, ln1g + l * Dd, ln1b + l * Dd, h16);
    gemm_kernel<false><<<dim3(Dd / BN, BT / BM), blk, 0, stream>>>(
        h16, Dd, Wq + (size_t)l * Dd * Dd, Dd, nullptr, nullptr, nullptr, q16, Dd, BT, Dd, Dd);
    gemm_kernel<false><<<dim3(Dd / BN, BT / BM), blk, 0, stream>>>(
        h16, Dd, Wk + (size_t)l * Dd * Dd, Dd, nullptr, nullptr, nullptr, k16, Dd, BT, Dd, Dd);
    gemm_kernel<false><<<dim3(Dd / BN, BT / BM), blk, 0, stream>>>(
        h16, Dd, Wvv + (size_t)l * Dd * Dd, Dd, nullptr, nullptr, nullptr, v16, Dd, BT, Dd, Dd);
    // fused causal attention: q,k,v -> attn (no materialized scores)
    flash_attn_kernel<<<dim3(Tt / 64, Bb * Hh), dim3(128), 0, stream>>>(
        q16, k16, v16, attn16, scale);
    // x = x + attn @ Wo + bo
    gemm_kernel<false><<<dim3(Dd / BN, BT / BM), blk, 0, stream>>>(
        attn16, Dd, Wo + (size_t)l * Dd * Dd, Dd, bo + l * Dd, x, x, nullptr, Dd, BT, Dd, Dd);
    layernorm_kernel<<<BT, blk, 0, stream>>>(x, ln2g + l * Dd, ln2b + l * Dd, h16);
    // mlp = relu(h @ W1 + b1)
    gemm_kernel<true><<<dim3(FF / BN, BT / BM), blk, 0, stream>>>(
        h16, Dd, W1 + (size_t)l * Dd * FF, FF, b1 + l * FF, nullptr, nullptr, mlp16, FF, BT, FF, Dd);
    // x = x + mlp @ W2 + b2
    gemm_kernel<false><<<dim3(Dd / BN, BT / BM), blk, 0, stream>>>(
        mlp16, FF, W2 + (size_t)l * FF * Dd, Dd, b2 + l * Dd, x, x, nullptr, Dd, BT, Dd, FF);
  }

  layernorm_kernel<<<BT, blk, 0, stream>>>(x, lnfg, lnfb, h16);
  // logits = h @ Whead + bhead  (bandwidth-dominant: 262 MB f32 out)
  gemm_kernel<false><<<dim3(Vv / BN, BT / BM), blk, 0, stream>>>(
      h16, Dd, Whead, Vv, bhead, nullptr, logits, nullptr, Vv, BT, Vv, Dd);

  zero_loss_kernel<<<1, 1, 0, stream>>>(loss);
  loss_kernel<<<BT, blk, 0, stream>>>(logits, tgts, loss);
}
